// MaskAttention_31353261261566
// MI455X (gfx1250) — matrix operand order repaired and checked by reading, not compiled, not run on hardware
//
#include <hip/hip_runtime.h>
#include <math.h>

// ---------------------------------------------------------------------------
// MaskAttention for MI455X (gfx1250, wave32, WMMA + Tensor Data Mover).
// Pipeline: qkv GEMM (bf16 WMMA) -> lepe dwconv -> flash attention (TDM-staged
// tiles in LDS, bf16 WMMA, fp32 online softmax) -> out projection (bf16 WMMA).
// ---------------------------------------------------------------------------

typedef __attribute__((ext_vector_type(16))) __bf16 v16bf;
typedef __attribute__((ext_vector_type(8)))  float  v8f;

#define B_    2
#define S_    3144
#define C_    256
#define H_    8
#define HD_   32
#define QN_   8
#define HH_   56
#define WW_   56
#define NPIX_ (HH_ * WW_)     // 3136
#define BS_   (B_ * S_)       // 6288
#define SCALE_ 0.17677669529663687f   // 32^-0.5

// --- WMMA 16-bit fragment K maps (per CDNA5 ISA 7.12.2) ---------------------
// A matrix (16x32, row per lane%16):
//   lanes 0-15 : K = {0..7} in v0-3, {16..23} in v4-7
//   lanes 16-31: K = {8..15},        {24..31}
__device__ __forceinline__ int kmapA(int e, int lane) {
  return ((e < 8) ? 0 : 16) + ((lane & 16) ? 8 : 0) + (e & 7);
}
// B matrix (32x16, column per lane%16):
//   lanes 0-15 : K = {0..7}, {16..23}; lanes 16-31: K = {8..15}, {24..31}
__device__ __forceinline__ int kmapB(int e, int lane) {
  return ((e < 8) ? 0 : 16) + ((lane & 16) ? 8 : 0) + (e & 7);
}

// Load a 16x32 fragment from a row-major fp32 matrix (convert to bf16).
template <bool IS_B>
__device__ __forceinline__ v16bf frag_f32(const float* __restrict__ p, int ld,
                                          int row, int kbase, int lane) {
  v16bf a;
  const float* r = p + (size_t)row * ld + kbase;
#pragma unroll
  for (int e = 0; e < 16; ++e) {
    int k = IS_B ? kmapB(e, lane) : kmapA(e, lane);
    a[e] = (__bf16)r[k];
  }
  return a;
}

// Load a 16x32 fragment from a row-major bf16 matrix.
template <bool IS_B>
__device__ __forceinline__ v16bf frag_bf16(const __bf16* __restrict__ p, int ld,
                                           int row, int kbase, int lane) {
  v16bf a;
  const __bf16* r = p + (size_t)row * ld + kbase;
#pragma unroll
  for (int e = 0; e < 16; ++e) {
    int k = IS_B ? kmapB(e, lane) : kmapA(e, lane);
    a[e] = r[k];
  }
  return a;
}

// ---------------------------------------------------------------------------
// Tensor Data Mover path.  This toolchain exposes the 6-arg builtin
// (uint32x4 g0, int32x8 g1, int32x4 g2, int32x4 g3, int32x8, i32 cpol).
// Builds a real 2D D# descriptor per CDNA5 ISA ch.8; unused groups zeroed.
// Guarded so a missing builtin falls back to manual LDS staging.
// ---------------------------------------------------------------------------
#if defined(__HIP_DEVICE_COMPILE__) && \
    __has_builtin(__builtin_amdgcn_tensor_load_to_lds) && \
    __has_builtin(__builtin_amdgcn_s_wait_tensorcnt)
#define USE_TDM 1
typedef __attribute__((ext_vector_type(4))) unsigned int u32x4;
typedef __attribute__((ext_vector_type(8))) int i32x8;
typedef __attribute__((ext_vector_type(4))) int i32x4;

// 2D tile DMA: global (row-major, stride0 elements between rows) -> LDS packed.
// OOB rows/cols (beyond tensor_d0/d1) read as zero (ISA 8.2), which also keeps
// all memory traffic in-bounds on the ragged S=3144 edges.
__device__ __forceinline__ void tdm_load_2d(unsigned lds_off, const void* gaddr,
                                            unsigned data_size_log2,
                                            unsigned tensor_d0, unsigned tensor_d1,
                                            unsigned tile_d0, unsigned tile_d1,
                                            unsigned long long stride0) {
  unsigned long long ga = (unsigned long long)(size_t)gaddr;
  u32x4 g0;
  g0.x = 1u;                                         // count=1 (valid), user mode
  g0.y = lds_off;                                    // lds_addr (bytes)
  g0.z = (unsigned)(ga & 0xFFFFFFFFu);               // global_addr[31:0]
  g0.w = (unsigned)((ga >> 32) & 0x01FFFFFFu)        // global_addr[56:32]
       | (2u << 30);                                 // type = 2 ("image")
  unsigned long long q0 = ((unsigned long long)(data_size_log2 & 3u) << 16)  // data_size
                        | ((unsigned long long)(tensor_d0 & 0xFFFFu) << 48); // dim0[15:0]
  unsigned long long q1 = ((unsigned long long)(tensor_d0 >> 16) & 0xFFFFu)  // dim0[31:16]
                        | ((unsigned long long)tensor_d1 << 16)              // dim1
                        | ((unsigned long long)(tile_d0 & 0xFFFFu) << 48);   // tile0
  unsigned long long q2 = ((unsigned long long)(tile_d1 & 0xFFFFu))          // tile1
                        | ((stride0 & 0xFFFFFFFFull) << 32);                 // stride0[31:0]
  unsigned long long q3 = ((stride0 >> 32) & 0xFFFFull);                     // stride0[47:32]
  i32x8 g1;
  g1[0] = (int)(unsigned)(q0 & 0xFFFFFFFFu); g1[1] = (int)(unsigned)(q0 >> 32);
  g1[2] = (int)(unsigned)(q1 & 0xFFFFFFFFu); g1[3] = (int)(unsigned)(q1 >> 32);
  g1[4] = (int)(unsigned)(q2 & 0xFFFFFFFFu); g1[5] = (int)(unsigned)(q2 >> 32);
  g1[6] = (int)(unsigned)(q3 & 0xFFFFFFFFu); g1[7] = (int)(unsigned)(q3 >> 32);
  i32x4 z4 = {0, 0, 0, 0};
  i32x8 z8 = {0, 0, 0, 0, 0, 0, 0, 0};
  __builtin_amdgcn_tensor_load_to_lds(g0, g1, z4, z4, z8, 0);
}
#endif

// ---------------------------------------------------------------------------
// Kernel 1: fused Q/K/V projection.  grid = 393 row-tiles, block = 128 (4 waves),
// each wave owns 64 output columns of all three matrices.
// ---------------------------------------------------------------------------
__global__ void qkv_kernel(const float* __restrict__ x,
                           const float* __restrict__ wq, const float* __restrict__ bq,
                           const float* __restrict__ wk, const float* __restrict__ bk,
                           const float* __restrict__ wv, const float* __restrict__ bv,
                           __bf16* __restrict__ qbf, __bf16* __restrict__ kbf,
                           float* __restrict__ vf) {
  const int lane = threadIdx.x & 31;
  const int wave = threadIdx.x >> 5;
  const int rowbase = blockIdx.x * 16;
  const int arow = rowbase + (lane & 15);

  const v8f zero8 = {0.f, 0.f, 0.f, 0.f, 0.f, 0.f, 0.f, 0.f};
  v8f aq[4], ak[4], av[4];
#pragma unroll
  for (int t = 0; t < 4; ++t) { aq[t] = zero8; ak[t] = zero8; av[t] = zero8; }

  for (int kb = 0; kb < C_; kb += 32) {
    v16bf ax = frag_f32<false>(x, C_, arow, kb, lane);
#pragma unroll
    for (int t = 0; t < 4; ++t) {
      const int nrow = wave * 64 + t * 16 + (lane & 15);
      v16bf bf;
      bf = frag_f32<true>(wq, C_, nrow, kb, lane);
      aq[t] = __builtin_amdgcn_wmma_f32_16x16x32_bf16(false, ax, false, bf,
                                                      (short)0, aq[t], false, false);
      bf = frag_f32<true>(wk, C_, nrow, kb, lane);
      ak[t] = __builtin_amdgcn_wmma_f32_16x16x32_bf16(false, ax, false, bf,
                                                      (short)0, ak[t], false, false);
      bf = frag_f32<true>(wv, C_, nrow, kb, lane);
      av[t] = __builtin_amdgcn_wmma_f32_16x16x32_bf16(false, ax, false, bf,
                                                      (short)0, av[t], false, false);
    }
  }

  const int hi8 = (lane & 16) ? 8 : 0;
#pragma unroll
  for (int t = 0; t < 4; ++t) {
    const int n = wave * 64 + t * 16 + (lane & 15);
#pragma unroll
    for (int r = 0; r < 8; ++r) {
      const int m = rowbase + r + hi8;
      const size_t idx = (size_t)m * C_ + n;
      qbf[idx] = (__bf16)(aq[t][r] + bq[n]);
      kbf[idx] = (__bf16)((ak[t][r] + bk[n]) * SCALE_);  // scaled K == attention V
      vf[idx]  = av[t][r] + bv[n];
    }
  }
}

// ---------------------------------------------------------------------------
// Kernel 2: lepe 5x5 depthwise conv on v[:, QN:].  Tiny FLOP count -> VALU.
// ---------------------------------------------------------------------------
__global__ void lepe_kernel(const float* __restrict__ vf,
                            const float* __restrict__ lw,
                            const float* __restrict__ lb,
                            float* __restrict__ lepe) {
  const int b = blockIdx.x / NPIX_;
  const int pix = blockIdx.x - b * NPIX_;
  const int y = pix / WW_;
  const int xx = pix - y * WW_;
  const int c = threadIdx.x;
  float acc = lb[c];
#pragma unroll
  for (int ky = 0; ky < 5; ++ky) {
    const int iy = y + ky - 2;
    if (iy < 0 || iy >= HH_) continue;
#pragma unroll
    for (int kx = 0; kx < 5; ++kx) {
      const int ix = xx + kx - 2;
      if (ix < 0 || ix >= WW_) continue;
      acc += vf[((size_t)b * S_ + QN_ + iy * WW_ + ix) * C_ + c]
           * lw[(ky * 5 + kx) * C_ + c];
    }
  }
  lepe[((size_t)b * NPIX_ + pix) * C_ + c] = acc;
}

// ---------------------------------------------------------------------------
// Kernel 3: flash attention.  grid = (197 q-tiles, 8 heads, 2 batch), one
// wave32 per block.  Per 32-key step: TDM DMAs the 32x32 bf16 K/V tile and the
// 16x32 fp32 mask tile into LDS, then 2 QK^T WMMAs, fp32 online softmax
// (16-lane shfl reductions), LDS re-layout of P, and 2 P*V WMMAs.
// ---------------------------------------------------------------------------
__global__ void attn_kernel(const __bf16* __restrict__ qbf,
                            const __bf16* __restrict__ kbf,
                            const float* __restrict__ mask,
                            float* __restrict__ out) {
  __shared__ __align__(16) __bf16 kvt[32 * 32];   // packed [key][hd] tile (2 KB)
  __shared__ __align__(16) float  mt[16 * 32];    // packed [qrow][key] mask tile
  __shared__ float Pt[16][33];                    // P re-layout staging

  const int lane = threadIdx.x & 31;
  const int hi = (lane & 16) ? 1 : 0;
  const int n16 = lane & 15;
  const int qt = blockIdx.x, h = blockIdx.y, b = blockIdx.z;
  const int qbase = qt * 16;
  const int rowOff = b * S_;
  const int col0 = h * HD_;

  int qr = qbase + n16; if (qr >= S_) qr = S_ - 1;           // clamp padded rows
  const v16bf qf = frag_bf16<false>(qbf, C_, rowOff + qr, col0, lane);

  const v8f zero8 = {0.f, 0.f, 0.f, 0.f, 0.f, 0.f, 0.f, 0.f};
  v8f o0 = zero8, o1 = zero8;
  float m_[8], l_[8];
#pragma unroll
  for (int r = 0; r < 8; ++r) { m_[r] = -1e30f; l_[r] = 0.f; }

  const int nkt = (S_ + 31) / 32;   // 99 key-pair tiles
  for (int kt = 0; kt < nkt; ++kt) {
    const int keybase = kt * 32;
    __syncthreads();   // previous iteration's LDS reads done before overwrite

    // ---- stage K/V tile (32 keys x 32 hd, bf16) and mask tile into LDS ----
#ifdef USE_TDM
    tdm_load_2d((unsigned)(size_t)(void*)kvt,
                kbf + (size_t)(rowOff + keybase) * C_ + col0,
                /*log2 size=*/1,
                /*tensor_d0=*/HD_, /*tensor_d1=*/(unsigned)(S_ - keybase),
                /*tile_d0=*/HD_, /*tile_d1=*/32,
                /*stride0=*/(unsigned long long)C_);
    tdm_load_2d((unsigned)(size_t)(void*)mt,
                mask + ((size_t)b * S_ + qbase) * S_ + keybase,
                /*log2 size=*/2,
                /*tensor_d0=*/(unsigned)(S_ - keybase),
                /*tensor_d1=*/(unsigned)(S_ - qbase),
                /*tile_d0=*/32, /*tile_d1=*/16,
                /*stride0=*/(unsigned long long)S_);
    __builtin_amdgcn_s_wait_tensorcnt(0);
#else
    {
      int krow = keybase + lane; if (krow >= S_) krow = S_ - 1;
      const uint4* src = (const uint4*)(kbf + (size_t)(rowOff + krow) * C_ + col0);
      uint4* dst = (uint4*)(kvt + lane * 32);
      dst[0] = src[0]; dst[1] = src[1]; dst[2] = src[2]; dst[3] = src[3];
      const int mrow = lane & 15, half = lane >> 4;
      const int qrow = qbase + mrow;
      for (int j = 0; j < 16; ++j) {
        const int kcol = keybase + half * 16 + j;
        float mv = (qrow < S_ && kcol < S_)
                 ? mask[((size_t)b * S_ + qrow) * S_ + kcol] : 0.f;
        mt[mrow * 32 + half * 16 + j] = mv;
      }
    }
#endif
    __syncthreads();

    // K fragments (B layout: lane = key row, K = head dim) from LDS
    v16bf kf0, kf1;
#pragma unroll
    for (int e = 0; e < 16; ++e) {
      kf0[e] = kvt[n16 * 32 + kmapB(e, lane)];
      kf1[e] = kvt[(16 + n16) * 32 + kmapB(e, lane)];
    }
    v8f lg0 = __builtin_amdgcn_wmma_f32_16x16x32_bf16(false, qf, false, kf0,
                                                      (short)0, zero8, false, false);
    v8f lg1 = __builtin_amdgcn_wmma_f32_16x16x32_bf16(false, qf, false, kf1,
                                                      (short)0, zero8, false, false);

#pragma unroll
    for (int r = 0; r < 8; ++r) {
      const int qrow = qbase + r + hi * 8;
      const int k0 = keybase + n16, k1 = keybase + 16 + n16;
      float v0 = -1e30f, v1 = -1e30f;
      if (qrow < S_) {
        const int prow = r + hi * 8;
        if (k0 < S_) v0 = lg0[r] + __logf(mt[prow * 32 + n16] + 1e-6f);
        if (k1 < S_) v1 = lg1[r] + __logf(mt[prow * 32 + 16 + n16] + 1e-6f);
      }
      // row-max across the 16-lane half holding this row
      float t = fmaxf(v0, v1);
      t = fmaxf(t, __shfl_xor(t, 1, 16));
      t = fmaxf(t, __shfl_xor(t, 2, 16));
      t = fmaxf(t, __shfl_xor(t, 4, 16));
      t = fmaxf(t, __shfl_xor(t, 8, 16));
      const float mnew = fmaxf(m_[r], t);
      const float scale = __expf(m_[r] - mnew);
      const float p0 = __expf(v0 - mnew);
      const float p1 = __expf(v1 - mnew);
      float rs = p0 + p1;
      rs += __shfl_xor(rs, 1, 16);
      rs += __shfl_xor(rs, 2, 16);
      rs += __shfl_xor(rs, 4, 16);
      rs += __shfl_xor(rs, 8, 16);
      l_[r] = l_[r] * scale + rs;
      m_[r] = mnew;
      o0[r] *= scale;
      o1[r] *= scale;
      const int prow = r + hi * 8;
      Pt[prow][n16] = p0;
      Pt[prow][16 + n16] = p1;
    }
    __syncthreads();

    // P as bf16 A-fragment; V column fragments from the LDS tile
    v16bf pf, vb0, vb1;
#pragma unroll
    for (int e = 0; e < 16; ++e) {
      pf[e] = (__bf16)Pt[n16][kmapA(e, lane)];
      const int kk = kmapB(e, lane);
      vb0[e] = kvt[kk * 32 + n16];
      vb1[e] = kvt[kk * 32 + 16 + n16];
    }
    o0 = __builtin_amdgcn_wmma_f32_16x16x32_bf16(false, pf, false, vb0,
                                                 (short)0, o0, false, false);
    o1 = __builtin_amdgcn_wmma_f32_16x16x32_bf16(false, pf, false, vb1,
                                                 (short)0, o1, false, false);
  }

#pragma unroll
  for (int r = 0; r < 8; ++r) {
    const int qrow = qbase + r + hi * 8;
    if (qrow < S_) {
      const float inv = 1.0f / l_[r];
      const size_t idx = ((size_t)rowOff + qrow) * C_ + col0;
      out[idx + n16] = o0[r] * inv;
      out[idx + 16 + n16] = o1[r] * inv;
    }
  }
}

// ---------------------------------------------------------------------------
// Kernel 4: output projection, fusing the +lepe residual into the A fragment.
// ---------------------------------------------------------------------------
__global__ void outproj_kernel(const float* __restrict__ att,
                               const float* __restrict__ lepe,
                               const float* __restrict__ wo,
                               const float* __restrict__ bo,
                               float* __restrict__ out) {
  const int lane = threadIdx.x & 31;
  const int wave = threadIdx.x >> 5;
  const int rowbase = blockIdx.x * 16;
  const int arow = rowbase + (lane & 15);
  const int bb = arow / S_;
  const int ss = arow - bb * S_;

  const v8f zero8 = {0.f, 0.f, 0.f, 0.f, 0.f, 0.f, 0.f, 0.f};
  v8f acc[4];
#pragma unroll
  for (int t = 0; t < 4; ++t) acc[t] = zero8;

  for (int kb = 0; kb < C_; kb += 32) {
    v16bf ax;
#pragma unroll
    for (int e = 0; e < 16; ++e) {
      const int k = kb + kmapA(e, lane);
      float g = att[(size_t)arow * C_ + k];
      if (ss >= QN_) g += lepe[((size_t)bb * NPIX_ + (ss - QN_)) * C_ + k];
      ax[e] = (__bf16)g;
    }
#pragma unroll
    for (int t = 0; t < 4; ++t) {
      const int nrow = wave * 64 + t * 16 + (lane & 15);
      const v16bf bf = frag_f32<true>(wo, C_, nrow, kb, lane);
      acc[t] = __builtin_amdgcn_wmma_f32_16x16x32_bf16(false, ax, false, bf,
                                                       (short)0, acc[t], false, false);
    }
  }

  const int hi8 = (lane & 16) ? 8 : 0;
#pragma unroll
  for (int t = 0; t < 4; ++t) {
    const int n = wave * 64 + t * 16 + (lane & 15);
#pragma unroll
    for (int r = 0; r < 8; ++r) {
      const int m = rowbase + r + hi8;
      out[(size_t)m * C_ + n] = acc[t][r] + bo[n];
    }
  }
}

// ---------------------------------------------------------------------------
extern "C" void kernel_launch(void* const* d_in, const int* in_sizes, int n_in,
                              void* d_out, int out_size, void* d_ws, size_t ws_size,
                              hipStream_t stream) {
  const float* x    = (const float*)d_in[0];
  const float* mask = (const float*)d_in[1];
  const float* wq   = (const float*)d_in[2];
  const float* bq   = (const float*)d_in[3];
  const float* wk   = (const float*)d_in[4];
  const float* bk   = (const float*)d_in[5];
  const float* wv   = (const float*)d_in[6];
  const float* bv   = (const float*)d_in[7];
  const float* lw   = (const float*)d_in[8];
  const float* lb   = (const float*)d_in[9];
  const float* wo   = (const float*)d_in[10];
  const float* bo   = (const float*)d_in[11];
  (void)in_sizes; (void)n_in; (void)out_size; (void)ws_size;

  // Workspace carve-out (~26 MB total)
  char* ws = (char*)d_ws;
  size_t off = 0;
  auto carve = [&](size_t bytes) -> char* {
    char* p = ws + off;
    off = (off + bytes + 255) & ~(size_t)255;
    return p;
  };
  __bf16* qbf = (__bf16*)carve((size_t)BS_ * C_ * sizeof(__bf16));
  __bf16* kbf = (__bf16*)carve((size_t)BS_ * C_ * sizeof(__bf16));
  float*  vf   = (float*)carve((size_t)BS_ * C_ * sizeof(float));
  float*  att  = (float*)carve((size_t)BS_ * C_ * sizeof(float));
  float*  lep  = (float*)carve((size_t)B_ * NPIX_ * C_ * sizeof(float));

  qkv_kernel<<<BS_ / 16, 128, 0, stream>>>(x, wq, bq, wk, bk, wv, bv, qbf, kbf, vf);
  lepe_kernel<<<B_ * NPIX_, C_, 0, stream>>>(vf, lw, lb, lep);
  dim3 ag((S_ + 15) / 16, H_, B_);
  attn_kernel<<<ag, 32, 0, stream>>>(qbf, kbf, mask, att);
  outproj_kernel<<<BS_ / 16, 128, 0, stream>>>(att, lep, wo, bo, (float*)d_out);
}
